// SelfAttnLayer_42116449305117
// MI455X (gfx1250) — matrix operand order repaired and checked by reading, not compiled, not run on hardware
//
#include <hip/hip_runtime.h>
#include <hip/hip_fp16.h>

#define NSEQ 2048
#define BATCH 4
#define DHID 512
#define NH 8
#define DH 64
#define DQ 256            // 4*DH = per-head attention feature dim
#define ATTN_FACTOR 0.0625f  // 0.5 / sqrt(64)

typedef __bf16 bf16;
typedef __attribute__((ext_vector_type(16))) __bf16 v16bf;
typedef __attribute__((ext_vector_type(8)))  __bf16 v8bf;
typedef __attribute__((ext_vector_type(8)))  float   v8f;

static __device__ __forceinline__ v8f wmma_bf16(v16bf a, v16bf b, v8f c) {
  // D = A(16x32) x B(32x16) + C, f32 accumulate
  return __builtin_amdgcn_wmma_f32_16x16x32_bf16(false, a, false, b, (short)0, c, false, false);
}

// A-matrix fragment: 16 rows x 32 K (16-bit). Row-major source, row stride ld (elements).
// K order used here (matches ldfragB): lane = M%16; lanes 0-15 hold K 0-15, lanes 16-31 hold K 16-31 (contiguous).
static __device__ __forceinline__ v16bf ldfragA(const bf16* base, int ld) {
  const int lane = threadIdx.x & 31;
  const bf16* p = base + (lane & 15) * ld + ((lane >> 4) & 1) * 16;
  union { v16bf v; v8bf h[2]; } u;
  u.h[0] = *(const v8bf*)p;
  u.h[1] = *(const v8bf*)(p + 8);
  return u.v;
}

// B-matrix fragment: 32 K x 16 cols. Source stored row-major as B^T: [16 rows=n][K], stride ld.
// K order used here (matches ldfragA): lane = N%16; lanes 0-15 hold K 0-15, lanes 16-31 hold K 16-31 (contiguous).
static __device__ __forceinline__ v16bf ldfragB(const bf16* base, int ld) {
  const int lane = threadIdx.x & 31;
  const bf16* p = base + (lane & 15) * ld + ((lane >> 4) & 1) * 16;
  return *(const v16bf*)p;
}

static __device__ __forceinline__ v8f vzero8() {
  v8f z;
#pragma unroll
  for (int i = 0; i < 8; ++i) z[i] = 0.0f;
  return z;
}

// ---------------- stage 0: conversions ----------------
__global__ void k_f2bf(const float* __restrict__ in, bf16* __restrict__ out, int n) {
  int i = blockIdx.x * blockDim.x + threadIdx.x;
  if (i < n) out[i] = (bf16)in[i];
}

// biasC[b][n][m] = rbf[0,b,0,n,m] + D[b,n,m]  (fp16), mask[b,m]==0 -> -65504
__global__ void k_bias(const float* __restrict__ rbf, const float* __restrict__ Dm,
                       const int* __restrict__ mask, __half* __restrict__ out) {
  int i = blockIdx.x * blockDim.x + threadIdx.x;  // [b][n][m], total 2^24
  int m = i & (NSEQ - 1);
  int b = i >> 22;                                // N*N = 2^22
  float v = rbf[i] + Dm[i];
  if (mask[b * NSEQ + m] == 0) v = -65504.0f;
  out[i] = __float2half(v);
}

// ---------------- stage 1: Q/K projections ----------------
// C[m,o] = sum_k Hb[m,k]*W[o,k] + bias[o], written to out[b][h=o/256][n][o%256] (bf16)
// grid (64, 32), block 256 (8 waves): block tile 128 rows x 64 cols, wave = 16x64.
__global__ void k_proj_qk(const bf16* __restrict__ A, const bf16* __restrict__ W,
                          const float* __restrict__ bias, bf16* __restrict__ out) {
  const int wave = threadIdx.x >> 5, lane = threadIdx.x & 31;
  const int row0 = blockIdx.x * 128 + wave * 16;
  const int col0 = blockIdx.y * 64;
  v8f acc[4] = {vzero8(), vzero8(), vzero8(), vzero8()};
  for (int k = 0; k < DHID; k += 32) {
    v16bf a = ldfragA(A + row0 * DHID + k, DHID);
#pragma unroll
    for (int t = 0; t < 4; ++t) {
      v16bf bw = ldfragB(W + (col0 + t * 16) * DHID + k, DHID);
      acc[t] = wmma_bf16(a, bw, acc[t]);
    }
  }
  const int c = lane & 15, rb = (lane >> 4) ? 8 : 0;
#pragma unroll
  for (int t = 0; t < 4; ++t) {
    int o = col0 + t * 16 + c;
    int h = o / DQ, j = o % DQ;
    float bi = bias[o];
#pragma unroll
    for (int g = 0; g < 8; ++g) {
      int m = row0 + rb + g;
      int b = m / NSEQ, n = m % NSEQ;
      out[((b * NH + h) * NSEQ + n) * DQ + j] = (bf16)(acc[t][g] + bi);
    }
  }
}

// ---------------- stage 1b: V projection, stored TRANSPOSED Vt[b][h][e][m] ----------------
// e<64: Wv row h*64+e over Hb rows (+bv). e>=64: c=(e-64)/64, Wvv row h*64+e%64 over Vb rows.
// Computed transposed: C[e,m] = sum_k W'[e,k]*Src[m,k]. grid (8, 8, 32), block 256.
__global__ void k_proj_v(const bf16* __restrict__ Hb, const bf16* __restrict__ Vb,
                         const bf16* __restrict__ Wv, const bf16* __restrict__ Wvv,
                         const float* __restrict__ bv, bf16* __restrict__ Vt) {
  const int wave = threadIdx.x >> 5, lane = threadIdx.x & 31;
  const int bh = blockIdx.z, b = bh / NH, h = bh % NH;
  const int e0 = blockIdx.y * 32 + (wave >> 2) * 16;   // within [0,256)
  const int m0 = blockIdx.x * 256 + (wave & 3) * 64;
  const int grp = e0 >> 6;                              // 0:Hv, 1..3:Vv channel grp-1
  const int wrow0 = h * DH + (e0 & (DH - 1));
  const bf16* Wsrc = (grp == 0) ? Wv : Wvv;
  v8f acc[4] = {vzero8(), vzero8(), vzero8(), vzero8()};
  for (int k = 0; k < DHID; k += 32) {
    v16bf a = ldfragA(Wsrc + wrow0 * DHID + k, DHID);
#pragma unroll
    for (int t = 0; t < 4; ++t) {
      const bf16* sb;
      int ld;
      if (grp == 0) { sb = Hb + (b * NSEQ + m0 + t * 16) * DHID + k; ld = DHID; }
      else { sb = Vb + ((b * NSEQ + m0 + t * 16) * 3 + (grp - 1)) * DHID + k; ld = 3 * DHID; }
      v16bf bs = ldfragB(sb, ld);
      acc[t] = wmma_bf16(a, bs, acc[t]);
    }
  }
  const int c = lane & 15, rb = (lane >> 4) ? 8 : 0;
#pragma unroll
  for (int g = 0; g < 8; ++g) {
    int e = e0 + rb + g;
    float bi = (grp == 0) ? bv[h * DH + e] : 0.0f;
#pragma unroll
    for (int t = 0; t < 4; ++t) {
      int m = m0 + t * 16 + c;
      Vt[(bh * DQ + e) * NSEQ + m] = (bf16)(acc[t][g] + bi);
    }
  }
}

// ---------------- stage 2: flash attention ----------------
// grid (N/64, B*NH), block 128 (4 waves, 16 query rows each).
__global__ void k_attn(const bf16* __restrict__ Q, const bf16* __restrict__ K,
                       const bf16* __restrict__ Vt, const __half* __restrict__ biasC,
                       bf16* __restrict__ res) {
  __shared__ __align__(32) bf16 stage[4][16][32];
  const int wave = threadIdx.x >> 5, lane = threadIdx.x & 31;
  const int bh = blockIdx.y, b = bh / NH;
  const int n0 = blockIdx.x * 64 + wave * 16;
  const bf16* Qb  = Q  + (bh * NSEQ + n0) * DQ;
  const bf16* Kb  = K  + bh * NSEQ * DQ;
  const bf16* Vtb = Vt + bh * DQ * NSEQ;
  const __half* biasb = biasC + ((size_t)b * NSEQ + n0) * NSEQ;

  v16bf qf[8];
#pragma unroll
  for (int kk = 0; kk < 8; ++kk) qf[kk] = ldfragA(Qb + kk * 32, DQ);
  v8f O[16];
#pragma unroll
  for (int t = 0; t < 16; ++t) O[t] = vzero8();
  float rm[8], rs[8];
#pragma unroll
  for (int g = 0; g < 8; ++g) { rm[g] = -1.0e30f; rs[g] = 0.0f; }

  const int c = lane & 15, rb = (lane >> 4) ? 8 : 0;

  for (int m0 = 0; m0 < NSEQ; m0 += 32) {
    v8f s0 = vzero8(), s1 = vzero8();
#pragma unroll
    for (int kk = 0; kk < 8; ++kk) {
      v16bf b0 = ldfragB(Kb + (m0)      * DQ + kk * 32, DQ);
      v16bf b1 = ldfragB(Kb + (m0 + 16) * DQ + kk * 32, DQ);
      s0 = wmma_bf16(qf[kk], b0, s0);
      s1 = wmma_bf16(qf[kk], b1, s1);
    }
    // scale + bias (lane holds column m0+c / m0+16+c, regs hold rows rb..rb+7)
    float p0[8], p1[8], scl[8];
#pragma unroll
    for (int g = 0; g < 8; ++g) {
      const __half* bp = biasb + (rb + g) * NSEQ + m0 + c;
      s0[g] = s0[g] * ATTN_FACTOR + __half2float(bp[0]);
      s1[g] = s1[g] * ATTN_FACTOR + __half2float(bp[16]);
    }
#pragma unroll
    for (int g = 0; g < 8; ++g) {
      float mx = fmaxf(s0[g], s1[g]);
#pragma unroll
      for (int d = 1; d < 16; d <<= 1) mx = fmaxf(mx, __shfl_xor(mx, d, 32));
      float nm = fmaxf(rm[g], mx);
      scl[g] = __expf(rm[g] - nm);
      rm[g] = nm;
      p0[g] = __expf(s0[g] - nm);
      p1[g] = __expf(s1[g] - nm);
      float sum = p0[g] + p1[g];
#pragma unroll
      for (int d = 1; d < 16; d <<= 1) sum += __shfl_xor(sum, d, 32);
      rs[g] = rs[g] * scl[g] + sum;
    }
#pragma unroll
    for (int t = 0; t < 16; ++t)
#pragma unroll
      for (int g = 0; g < 8; ++g) O[t][g] *= scl[g];
    // re-layout P: C-fragment -> A-fragment via per-wave LDS staging
#pragma unroll
    for (int g = 0; g < 8; ++g) {
      stage[wave][rb + g][c]      = (bf16)p0[g];
      stage[wave][rb + g][c + 16] = (bf16)p1[g];
    }
    asm volatile("s_wait_dscnt 0" ::: "memory");
    v16bf pf = ldfragA(&stage[wave][0][0], 32);
    asm volatile("" ::: "memory");
#pragma unroll
    for (int t = 0; t < 16; ++t) {
      v16bf vf = ldfragB(Vtb + (t * 16) * NSEQ + m0, NSEQ);
      O[t] = wmma_bf16(pf, vf, O[t]);
    }
    asm volatile("s_wait_dscnt 0" ::: "memory");
  }
#pragma unroll
  for (int g = 0; g < 8; ++g) rs[g] = 1.0f / rs[g];
#pragma unroll
  for (int t = 0; t < 16; ++t)
#pragma unroll
    for (int g = 0; g < 8; ++g)
      res[(bh * NSEQ + n0 + rb + g) * DQ + t * 16 + c] = (bf16)(O[t][g] * rs[g]);
}

// ---------------- stage 3: output projections ----------------
// tgt 0: H_o = H_res@Wo.T+bo ; tgt 1..3: V_o[:,:,c] = V_res[:,:,c]@Wvo.T
// res is [b][h][n][256]; K chunk k -> res[b][k/64][n][off + k%64]. grid (64, 8, 4), block 256.
__global__ void k_out(const bf16* __restrict__ res, const bf16* __restrict__ Wo,
                      const bf16* __restrict__ Wvo, const float* __restrict__ bo,
                      float* __restrict__ out) {
  const int wave = threadIdx.x >> 5, lane = threadIdx.x & 31;
  const int row0 = blockIdx.x * 128 + wave * 16;
  const int col0 = blockIdx.y * 64;
  const int tgt = blockIdx.z;
  const bf16* W = (tgt == 0) ? Wo : Wvo;
  const int off = (tgt == 0) ? 0 : DH + (tgt - 1) * DH;
  const int b = row0 / NSEQ, n0 = row0 % NSEQ;
  v8f acc[4] = {vzero8(), vzero8(), vzero8(), vzero8()};
  for (int k = 0; k < DHID; k += 32) {
    int h = k / DH, j = k % DH;
    v16bf a = ldfragA(res + ((b * NH + h) * NSEQ + n0) * DQ + off + j, DQ);
#pragma unroll
    for (int t = 0; t < 4; ++t) {
      v16bf bw = ldfragB(W + (col0 + t * 16) * DHID + k, DHID);
      acc[t] = wmma_bf16(a, bw, acc[t]);
    }
  }
  const int c = lane & 15, rb = (lane >> 4) ? 8 : 0;
#pragma unroll
  for (int t = 0; t < 4; ++t) {
    int o = col0 + t * 16 + c;
#pragma unroll
    for (int g = 0; g < 8; ++g) {
      long m = row0 + rb + g;  // = b*N + n
      if (tgt == 0) out[m * DHID + o] = acc[t][g] + bo[o];
      else out[(long)BATCH * NSEQ * DHID + (m * 3 + (tgt - 1)) * DHID + o] = acc[t][g];
    }
  }
}

extern "C" void kernel_launch(void* const* d_in, const int* in_sizes, int n_in,
                              void* d_out, int out_size, void* d_ws, size_t ws_size,
                              hipStream_t stream) {
  (void)in_sizes; (void)n_in; (void)out_size; (void)ws_size;
  const float* H   = (const float*)d_in[0];
  const float* V   = (const float*)d_in[1];
  const float* Dm  = (const float*)d_in[2];
  const float* rbf = (const float*)d_in[3];
  const int*   mask= (const int*)d_in[4];
  const float* Wq  = (const float*)d_in[5];
  const float* bq  = (const float*)d_in[6];
  const float* Wk  = (const float*)d_in[7];
  const float* bk  = (const float*)d_in[8];
  const float* Wv  = (const float*)d_in[9];
  const float* bv  = (const float*)d_in[10];
  const float* Wvv = (const float*)d_in[11];
  const float* Wo  = (const float*)d_in[12];
  const float* bo  = (const float*)d_in[13];
  const float* Wvo = (const float*)d_in[14];
  float* out = (float*)d_out;

  char* ws = (char*)d_ws;
  auto take = [&](size_t bytes) -> char* {
    char* p = ws; ws += (bytes + 255) & ~(size_t)255; return p;
  };
  const size_t M = (size_t)BATCH * NSEQ;  // 8192
  bf16* Hb   = (bf16*)take(M * DHID * 2);
  bf16* Vb   = (bf16*)take(M * 3 * DHID * 2);
  bf16* Wqb  = (bf16*)take((size_t)4 * DHID * DHID * 2);
  bf16* Wkb  = (bf16*)take((size_t)4 * DHID * DHID * 2);
  bf16* Wvb  = (bf16*)take((size_t)DHID * DHID * 2);
  bf16* Wvvb = (bf16*)take((size_t)DHID * DHID * 2);
  bf16* Wob  = (bf16*)take((size_t)DHID * DHID * 2);
  bf16* Wvob = (bf16*)take((size_t)DHID * DHID * 2);
  bf16* Qw   = (bf16*)take(M * DQ * 2);   // [b][h][n][256]
  bf16* Kw   = (bf16*)take(M * DQ * 2);   // [b][h][m][256]
  bf16* Vtw  = (bf16*)take(M * DQ * 2);   // [b][h][e][m]
  bf16* Resw = (bf16*)take(M * DQ * 2);   // [b][h][n][256]
  __half* BiasC = (__half*)take((size_t)BATCH * NSEQ * NSEQ * 2);

  auto cvt = [&](const float* src, bf16* dst, int n) {
    k_f2bf<<<dim3((n + 255) / 256), dim3(256), 0, stream>>>(src, dst, n);
  };
  cvt(H,   Hb,   (int)(M * DHID));
  cvt(V,   Vb,   (int)(M * 3 * DHID));
  cvt(Wq,  Wqb,  4 * DHID * DHID);
  cvt(Wk,  Wkb,  4 * DHID * DHID);
  cvt(Wv,  Wvb,  DHID * DHID);
  cvt(Wvv, Wvvb, DHID * DHID);
  cvt(Wo,  Wob,  DHID * DHID);
  cvt(Wvo, Wvob, DHID * DHID);

  k_bias<<<dim3(BATCH * NSEQ * NSEQ / 256), dim3(256), 0, stream>>>(rbf, Dm, mask, BiasC);

  k_proj_qk<<<dim3(64, 32), dim3(256), 0, stream>>>(Hb, Wqb, bq, Qw);
  k_proj_qk<<<dim3(64, 32), dim3(256), 0, stream>>>(Hb, Wkb, bk, Kw);
  k_proj_v <<<dim3(8, 8, 32), dim3(256), 0, stream>>>(Hb, Vb, Wvb, Wvvb, bv, Vtw);
  k_attn   <<<dim3(NSEQ / 64, BATCH * NH), dim3(128), 0, stream>>>(Qw, Kw, Vtw, BiasC, Resw);
  k_out    <<<dim3(64, 8, 4), dim3(256), 0, stream>>>(Resw, Wob, Wvob, bo, out);
}
